// DipNetEncoder_9766755631154
// MI455X (gfx1250) — matrix operand order repaired and checked by reading, not compiled, not run on hardware
//
#include <hip/hip_runtime.h>
#include <hip/hip_bf16.h>

// ---------------------------------------------------------------------------
// DipNet encoder for MI455X (gfx1250), compile-only tuned.
// bf16 storage everywhere (halves HBM/L2 traffic; activations fit 192MB L2),
// v_wmma_f32_16x16x32_bf16 for both GEMMs, fp32 BN/FiLM math, fp32 output.
// Activation buffers padded to stride 128 (zeroed tails) so WMMA A-tiles are
// unmasked 16B vector loads; W pre-packed per block into the WMMA B layout.
// BN statistics: column-major f32 LDS spill + linear reduction (no bpermute
// chains), fully deterministic.
// ---------------------------------------------------------------------------

typedef __attribute__((ext_vector_type(16))) __bf16 v16bf;
typedef __attribute__((ext_vector_type(8)))  __bf16 v8bf;
typedef __attribute__((ext_vector_type(8)))  float  v8f;

#define BN_EPS 1e-5f
#define B_    4096
#define N_    81
#define OUT_  120
#define MOD_  80
#define NPAD  96         // 81 -> 6x16 row tiles (3x32 K chunks)
#define OPAD  128        // 120 -> 8x16 col tiles; also activation row stride
#define PADR  100        // sY column-major row-dim pad (96 -> 100, x4B)
#define BNROWS (B_ * N_) // 331776, divisible by 16

// LDS layout for gcn_a (dynamic): sT 24576B | sA 18432B | sY 51200B = 94208B
#define GCNA_SMEM (NPAD * OPAD * 2 + NPAD * NPAD * 2 + OPAD * PADR * 4)

// ---------------------------------------------------------------------------
// embeddings  mod[b,0:60]=xp@pW+pb, mod[b,60:80]=xs@sW+sb
// ---------------------------------------------------------------------------
__global__ __launch_bounds__(256)
void embed_kernel(const float* __restrict__ xp, const float* __restrict__ xs,
                  const float* __restrict__ pW, const float* __restrict__ pb,
                  const float* __restrict__ sW, const float* __restrict__ sb,
                  float* __restrict__ mod)
{
    int idx = blockIdx.x * 256 + threadIdx.x;     // b*80 + k
    if (idx >= B_ * MOD_) return;
    int b = idx / MOD_, k = idx % MOD_;
    float v;
    if (k < 60) {
        v = pb[k];
        #pragma unroll
        for (int j = 0; j < 7; ++j) v += xp[b * 7 + j] * pW[j * 60 + k];
    } else {
        int kk = k - 60;
        v = sb[kk];
        #pragma unroll
        for (int j = 0; j < 3; ++j) v += xs[b * 3 + j] * sW[j * 20 + kk];
    }
    mod[idx] = v;
}

// ---------------------------------------------------------------------------
// FiLM  gamma/beta[b,o] = mod[b,:]@Wg/Wb + bg/bb   (K=80, tiny)
// ---------------------------------------------------------------------------
__global__ __launch_bounds__(256)
void film_kernel(const float* __restrict__ mod,
                 const float* __restrict__ Wg, const float* __restrict__ bg,
                 const float* __restrict__ Wb, const float* __restrict__ bb,
                 float* __restrict__ gamma, float* __restrict__ beta)
{
    int idx = blockIdx.x * 256 + threadIdx.x;     // b*120 + o
    if (idx >= B_ * OUT_) return;
    int b = idx / OUT_, o = idx % OUT_;
    float g = bg[o], be = bb[o];
    const float* m = mod + (size_t)b * MOD_;
    #pragma unroll 4
    for (int k = 0; k < MOD_; ++k) {
        float mv = m[k];
        g  += mv * Wg[k * OUT_ + o];
        be += mv * Wb[k * OUT_ + o];
    }
    gamma[idx] = g;
    beta[idx]  = be;
}

// ---------------------------------------------------------------------------
// Pack W (in x 120 f32, row-major) -> Wp[col][K] bf16, zero-padded to 128x128.
// K is contiguous per column, so gcn_w
// B-tiles become two aligned 16B vector loads.
// ---------------------------------------------------------------------------
__global__ __launch_bounds__(256)
void pack_w_kernel(const float* __restrict__ W, __bf16* __restrict__ Wp, int in)
{
    int idx = blockIdx.x * 256 + threadIdx.x;     // col*128 + K
    if (idx >= 128 * 128) return;
    int col = idx >> 7, K = idx & 127;
    float v = (col < OUT_ && K < in) ? W[K * OUT_ + col] : 0.0f;
    Wp[idx] = (__bf16)v;
}

// ---------------------------------------------------------------------------
// T = x @ W + bias  (M=331776, K=IN, N=120 pad 128), bf16 WMMA.
// One wave owns a 16(M) x 128(N) strip. CDNA5 16-bit operand layouts:
//   A 16x32: lane r / r+16 holds row r; elem e -> K = e + (e<8?0:8) + half*8
//   B 32x16: lane holds col lane&15;    elem e -> K = e + (e<8?0:8) + half*8
//   D 16x16: elem d -> row d + half*8, col lane&15
// bf16 path: x rows are stride-128 zero-padded -> unmasked 16B vector loads.
// ---------------------------------------------------------------------------
template <bool XF32, int IN>
__global__ __launch_bounds__(128)
void gcn_w_kernel(const void* __restrict__ xv, const __bf16* __restrict__ Wp,
                  const float* __restrict__ bias, __bf16* __restrict__ T)
{
    const int wave = blockIdx.x * (blockDim.x >> 5) + (threadIdx.x >> 5);
    const int lane = threadIdx.x & 31;
    const int r    = lane & 15;
    const int half = lane >> 4;
    const int row0 = wave * 16;
    if (row0 >= BNROWS) return;
    const int row = row0 + r;

    v8f c[8] = {};
    constexpr int NKC = (IN + 31) / 32;

    #pragma unroll
    for (int kc = 0; kc < NKC; ++kc) {
        const int k0 = kc * 32;
        v16bf a;
        if (XF32) {
            // first block: f32 input, stride IN, masked scalar loads
            const float* x = (const float*)xv;
            #pragma unroll
            for (int e = 0; e < 16; ++e) {
                int kk = k0 + half * 8 + e + ((e < 8) ? 0 : 8);
                a[e] = (__bf16)((kk < IN) ? x[(size_t)row * IN + kk] : 0.0f);
            }
        } else {
            // bf16 input, stride 128, zero-padded tail: unmasked vectors
            const __bf16* x  = (const __bf16*)xv;
            const __bf16* xr = x + (size_t)row * OPAD + k0 + half * 8;
            v8bf a0 = *(const v8bf*)(xr);
            v8bf a1 = *(const v8bf*)(xr + 16);
            a = __builtin_shufflevector(a0, a1, 0,1,2,3,4,5,6,7,8,9,10,11,12,13,14,15);
        }
        #pragma unroll
        for (int ct = 0; ct < 8; ++ct) {
            const int col = ct * 16 + r;
            const __bf16* wr = Wp + (size_t)col * 128 + k0 + half * 8;
            v8bf b0 = *(const v8bf*)(wr);
            v8bf b1 = *(const v8bf*)(wr + 16);
            v16bf bm = __builtin_shufflevector(b0, b1, 0,1,2,3,4,5,6,7,8,9,10,11,12,13,14,15);
            c[ct] = __builtin_amdgcn_wmma_f32_16x16x32_bf16(
                        false, a, false, bm, (short)0, c[ct], false, false);
        }
    }

    // bias + store T (bf16, stride 120)
    #pragma unroll
    for (int ct = 0; ct < 8; ++ct) {
        const int col = ct * 16 + r;
        if (col >= OUT_) continue;
        const float bv = bias[col];
        #pragma unroll
        for (int d = 0; d < 8; ++d) {
            int rr = row0 + d + half * 8;
            T[(size_t)rr * OUT_ + col] = (__bf16)(c[ct][d] + bv);
        }
    }
}

// ---------------------------------------------------------------------------
// y[b] = Apad @ T[b]  (96x96 @ 96x128, one WG per batch), fused BN partials.
// D tiles spill to column-major f32 LDS (rows contiguous per lane ->
// ds_store_b128 pairs). Stats: thread-per-row linear sum (conflict-free,
// deterministic). y written back cooperatively as packed-bf16 b32 stores.
// ---------------------------------------------------------------------------
__global__ __launch_bounds__(256)
void gcn_a_kernel(const __bf16* __restrict__ T, const float* __restrict__ Adj,
                  __bf16* __restrict__ Y,
                  float* __restrict__ psum, float* __restrict__ pssq)
{
    extern __shared__ char smem[];
    __bf16* sT = (__bf16*)smem;                                // [K=96][O=128]
    __bf16* sA = (__bf16*)(smem + NPAD * OPAD * 2);            // [n=96][m=96]
    float*  sY = (float*)(smem + NPAD * OPAD * 2 + NPAD * NPAD * 2); // [col=128][row pad 100]

    const int b   = blockIdx.x;
    const int tid = threadIdx.x;

    // stage T[b] (m<81, cols<120) as u32 copies; zero-pad the rest
    unsigned int*       sT32 = (unsigned int*)sT;
    const unsigned int* T32  = (const unsigned int*)(T + (size_t)b * N_ * OUT_);
    for (int idx = tid; idx < NPAD * 64; idx += 256) {
        int m = idx >> 6, cc = idx & 63;
        unsigned int v = 0u;
        if (m < N_ && cc < 60) v = T32[m * 60 + cc];
        sT32[m * 64 + cc] = v;
    }
    // stage adjacency (0/1 exact in bf16), zero-padded
    for (int idx = tid; idx < NPAD * NPAD; idx += 256) {
        int n = idx / NPAD, m = idx % NPAD;
        float v = (n < N_ && m < N_) ? Adj[n * N_ + m] : 0.0f;
        sA[idx] = (__bf16)v;
    }
    __syncthreads();

    const int wv   = tid >> 5;          // o-tile 0..7
    const int lane = tid & 31;
    const int r    = lane & 15;
    const int half = lane >> 4;
    const int col  = wv * 16 + r;

    for (int nt = 0; nt < 6; ++nt) {
        v8f c = {};
        #pragma unroll
        for (int kc = 0; kc < 3; ++kc) {
            v16bf a, bm;
            #pragma unroll
            for (int e = 0; e < 16; ++e) {
                int kk = kc * 32 + half * 8 + e + ((e < 8) ? 0 : 8);
                a[e] = sA[(nt * 16 + r) * NPAD + kk];
            }
            #pragma unroll
            for (int e = 0; e < 16; ++e) {
                int kk = kc * 32 + half * 8 + e + ((e < 8) ? 0 : 8);
                bm[e] = sT[kk * OPAD + col];
            }
            c = __builtin_amdgcn_wmma_f32_16x16x32_bf16(
                    false, a, false, bm, (short)0, c, false, false);
        }
        // spill: rows nt*16 + half*8 + (0..7) contiguous -> 2x b128
        float* dst = sY + (size_t)col * PADR + nt * 16 + half * 8;
        #pragma unroll
        for (int d = 0; d < 8; ++d) dst[d] = c[d];
    }
    __syncthreads();

    // BN partials: thread-per-row, conflict-free sequential sum over 128 cols
    if (tid < NPAD) {
        float s = 0.0f, q = 0.0f;
        #pragma unroll 8
        for (int cc = 0; cc < OPAD; ++cc) {
            float v = sY[(size_t)cc * PADR + tid];
            s += v; q += v * v;
        }
        psum[(size_t)b * NPAD + tid] = s;
        pssq[(size_t)b * NPAD + tid] = q;
    }

    // y write-back: packed bf16 pairs, coalesced b32 stores (stride-128 rows)
    unsigned int* Y32 = (unsigned int*)Y;
    for (int idx = tid; idx < N_ * 60; idx += 256) {
        int n = idx / 60, c2 = idx % 60;
        float f0 = sY[(size_t)(2 * c2)     * PADR + n];
        float f1 = sY[(size_t)(2 * c2 + 1) * PADR + n];
        union { unsigned int u; __bf16 h[2]; } p;
        p.h[0] = (__bf16)f0; p.h[1] = (__bf16)f1;
        Y32[((size_t)b * N_ + n) * (OPAD / 2) + c2] = p.u;
    }
}

// ---------------------------------------------------------------------------
// BN finalize -> scale[n]=g*rstd, shift[n]=bn_b - mean*scale
// ---------------------------------------------------------------------------
__global__ __launch_bounds__(256)
void bn_finalize_kernel(const float* __restrict__ psum, const float* __restrict__ pssq,
                        const float* __restrict__ bn_g, const float* __restrict__ bn_b,
                        float* __restrict__ ss)
{
    __shared__ float red[2][256];
    const int n   = blockIdx.x;        // 0..95 (81..95 harmless)
    const int tid = threadIdx.x;
    float s = 0.0f, q = 0.0f;
    for (int b = tid; b < B_; b += 256) {
        s += psum[(size_t)b * NPAD + n];
        q += pssq[(size_t)b * NPAD + n];
    }
    red[0][tid] = s; red[1][tid] = q;
    __syncthreads();
    for (int st = 128; st > 0; st >>= 1) {
        if (tid < st) { red[0][tid] += red[0][tid + st]; red[1][tid] += red[1][tid + st]; }
        __syncthreads();
    }
    if (tid == 0) {
        const float cnt  = (float)B_ * (float)OUT_;
        const float mean = red[0][0] / cnt;
        const float var  = red[1][0] / cnt - mean * mean;
        const float rstd = rsqrtf(var + BN_EPS);
        const float g  = (n < N_) ? bn_g[n] : 0.0f;
        const float bb = (n < N_) ? bn_b[n] : 0.0f;
        const float scale = g * rstd;
        ss[n]        = scale;
        ss[NPAD + n] = bb - mean * scale;
    }
}

// ---------------------------------------------------------------------------
// apply  out = relu(gamma*(y*scale+shift)+beta) [+ x], 2 elems/thread.
// LAST writes fp32 pairs into the concatenated (B,81,240) output.
// ---------------------------------------------------------------------------
template <bool RES, bool LAST>
__global__ __launch_bounds__(256)
void apply_kernel(const __bf16* __restrict__ Y, const __bf16* __restrict__ X,
                  const float* __restrict__ gamma, const float* __restrict__ beta,
                  const float* __restrict__ ss,
                  __bf16* __restrict__ Onext, float* __restrict__ Ofinal, int finoff)
{
    size_t idx = (size_t)blockIdx.x * 256 + threadIdx.x;   // (b*81+n)*60 + c2
    if (idx >= (size_t)BNROWS * 60) return;
    const int    c2   = (int)(idx % 60);
    const size_t rowi = idx / 60;                          // b*81 + n
    const int    n    = (int)(rowi % N_);
    const int    b    = (int)(rowi / N_);
    const int    o    = c2 * 2;

    union { unsigned int u; __bf16 h[2]; } uy;
    uy.u = ((const unsigned int*)Y)[rowi * (OPAD / 2) + c2];
    const float sc = ss[n], sh = ss[NPAD + n];
    const float g0 = gamma[b * OUT_ + o], g1 = gamma[b * OUT_ + o + 1];
    const float e0 = beta [b * OUT_ + o], e1 = beta [b * OUT_ + o + 1];
    float v0 = fmaxf(g0 * ((float)uy.h[0] * sc + sh) + e0, 0.0f);
    float v1 = fmaxf(g1 * ((float)uy.h[1] * sc + sh) + e1, 0.0f);
    if (RES) {
        union { unsigned int u; __bf16 h[2]; } ux;
        ux.u = ((const unsigned int*)X)[rowi * (OPAD / 2) + c2];
        v0 += (float)ux.h[0]; v1 += (float)ux.h[1];
    }
    if (LAST) {
        float* op = Ofinal + rowi * 240 + finoff + o;
        op[0] = v0; op[1] = v1;
    } else {
        union { unsigned int u; __bf16 h[2]; } uo;
        uo.h[0] = (__bf16)v0; uo.h[1] = (__bf16)v1;
        ((unsigned int*)Onext)[rowi * (OPAD / 2) + c2] = uo.u;
    }
}

// ---------------------------------------------------------------------------
// Host launch
// ---------------------------------------------------------------------------
extern "C" void kernel_launch(void* const* d_in, const int* in_sizes, int n_in,
                              void* d_out, int out_size, void* d_ws, size_t ws_size,
                              hipStream_t stream)
{
    (void)in_sizes; (void)n_in; (void)out_size; (void)ws_size;

    // setup_inputs() order:
    // 0 x_bo, 1 x_po, 2 x_power_1h, 3 x_season_1h, 4 A,
    // 5 power_W, 6 power_b, 7 season_W, 8 season_b,
    // 9..136 board_params[16] x {W,b,bn_g,bn_b,Wg,bg,Wb,bb}, 137..264 po_params
    const float* xp  = (const float*)d_in[2];
    const float* xs  = (const float*)d_in[3];
    const float* Adj = (const float*)d_in[4];
    const float* pW  = (const float*)d_in[5];
    const float* pb  = (const float*)d_in[6];
    const float* sW  = (const float*)d_in[7];
    const float* sb  = (const float*)d_in[8];

    char*  ws  = (char*)d_ws;
    size_t off = 0;
    auto alloc = [&](size_t bytes) -> void* {
        void* p = ws + off; off += (bytes + 255) & ~(size_t)255; return p;
    };
    const size_t bufBytes = (size_t)BNROWS * OPAD * 2;          // padded bf16 acts
    __bf16* T    = (__bf16*)alloc((size_t)BNROWS * OUT_ * 2);   // 79.6 MB
    __bf16* bufA = (__bf16*)alloc(bufBytes);                    // 85 MB
    __bf16* bufB = (__bf16*)alloc(bufBytes);                    // 85 MB
    __bf16* Wp   = (__bf16*)alloc((size_t)128 * 128 * 2);
    float*  mod   = (float*)alloc((size_t)B_ * MOD_ * 4);
    float*  gamma = (float*)alloc((size_t)B_ * OUT_ * 4);
    float*  beta  = (float*)alloc((size_t)B_ * OUT_ * 4);
    float*  psum  = (float*)alloc((size_t)B_ * NPAD * 4);
    float*  pssq  = (float*)alloc((size_t)B_ * NPAD * 4);
    float*  ss    = (float*)alloc((size_t)2 * NPAD * 4);

    // zero padded activation buffers (pad cols must read as 0 every call)
    hipMemsetAsync(bufA, 0, bufBytes, stream);
    hipMemsetAsync(bufB, 0, bufBytes, stream);

    embed_kernel<<<(B_ * MOD_ + 255) / 256, 256, 0, stream>>>(xp, xs, pW, pb, sW, sb, mod);

    const int gw_blocks    = (BNROWS / 16) / 4;                       // 4 waves/WG
    const int apply_blocks = (int)(((size_t)BNROWS * 60 + 255) / 256);
    const int film_blocks  = (B_ * OUT_ + 255) / 256;
    const int packw_blocks = (128 * 128) / 256;

    for (int branch = 0; branch < 2; ++branch) {
        const int   base   = 9 + branch * 128;
        const void* x0     = (branch == 0) ? d_in[0] : d_in[1];
        const int   in0    = (branch == 0) ? 35 : 40;
        const int   finoff = (branch == 0) ? 0 : 120;

        const __bf16* cur = nullptr;
        for (int i = 0; i < 16; ++i) {
            const float* W    = (const float*)d_in[base + i * 8 + 0];
            const float* bi   = (const float*)d_in[base + i * 8 + 1];
            const float* bn_g = (const float*)d_in[base + i * 8 + 2];
            const float* bn_b = (const float*)d_in[base + i * 8 + 3];
            const float* Wg   = (const float*)d_in[base + i * 8 + 4];
            const float* bg   = (const float*)d_in[base + i * 8 + 5];
            const float* Wb   = (const float*)d_in[base + i * 8 + 6];
            const float* bbf  = (const float*)d_in[base + i * 8 + 7];

            film_kernel<<<film_blocks, 256, 0, stream>>>(mod, Wg, bg, Wb, bbf, gamma, beta);
            pack_w_kernel<<<packw_blocks, 256, 0, stream>>>(W, Wp, (i == 0) ? in0 : OUT_);

            if (i == 0) {
                if (in0 == 35)
                    gcn_w_kernel<true, 35><<<gw_blocks, 128, 0, stream>>>(x0, Wp, bi, T);
                else
                    gcn_w_kernel<true, 40><<<gw_blocks, 128, 0, stream>>>(x0, Wp, bi, T);
            } else {
                gcn_w_kernel<false, 120><<<gw_blocks, 128, 0, stream>>>(cur, Wp, bi, T);
            }

            __bf16* nxt = (i & 1) ? bufB : bufA;
            gcn_a_kernel<<<B_, 256, GCNA_SMEM, stream>>>(T, Adj, nxt, psum, pssq);
            bn_finalize_kernel<<<NPAD, 256, 0, stream>>>(psum, pssq, bn_g, bn_b, ss);

            const __bf16* xres = (i == 0) ? nxt : cur;   // unused when !RES
            if (i == 0)
                apply_kernel<false, false><<<apply_blocks, 256, 0, stream>>>(
                    nxt, xres, gamma, beta, ss, nxt, (float*)d_out, finoff);
            else if (i < 15)
                apply_kernel<true, false><<<apply_blocks, 256, 0, stream>>>(
                    nxt, xres, gamma, beta, ss, nxt, (float*)d_out, finoff);
            else
                apply_kernel<true, true ><<<apply_blocks, 256, 0, stream>>>(
                    nxt, xres, gamma, beta, ss, nxt, (float*)d_out, finoff);

            cur = nxt;
        }
    }
}
